// SelfAttention2d_72318659330165
// MI455X (gfx1250) — compile-verified, matrix operand order checked by reading, not run on hardware
//
#include <hip/hip_runtime.h>
#include <hip/hip_bf16.h>
#include <stdint.h>

// ---------------------------------------------------------------------------
// SelfAttention2d for MI455X (gfx1250, wave32, WMMA + TDM).
//
// Roofline: ~40 GFLOP total, ~16 MB essential HBM traffic -> compute bound on
// the attention GEMMs. Flash attention keeps 4096x4096 scores out of HBM.
// bf16 WMMA (16x16x32, f32 accum) = 8x the K depth per issue of fp32 WMMA
// (16x16x4) with fp32 exponent range; head dim d=32 == one WMMA K step.
// K/V tiles are staged in LDS by the Tensor Data Mover (tensor_load_to_lds,
// double buffered, s_wait_tensorcnt) and shared by 8 waves (128 queries) per
// workgroup, cutting K/V re-read traffic 8x vs one-wave-per-tile.
// ---------------------------------------------------------------------------

typedef __attribute__((ext_vector_type(16))) __bf16 v16bf;
typedef __attribute__((ext_vector_type(8)))  float  v8f;
typedef __attribute__((ext_vector_type(4)))  unsigned int v4u;
typedef __attribute__((ext_vector_type(8)))  int v8i;
typedef __attribute__((ext_vector_type(4)))  int v4i;

#define BATCH 2
#define CCH   256
#define NPIX  4096
#define NH    8
#define DH    32
#define KCH   64     // keys per LDS-staged chunk

// CDNA5 16-bit A/B operand register layout (CDNA5 ISA, 16-bit A 16x32):
// half h (0..15) of the v16bf in a lane maps to K index:
//   k = 16*(h/8) + 8*(lane>=16) + (h&7)
__device__ __forceinline__ int kidx16(int h, int hi) {
  return ((h >> 3) << 4) + (hi << 3) + (h & 7);
}

__device__ __forceinline__ v8f wmma_bf16(v16bf a, v16bf b, v8f c) {
  return __builtin_amdgcn_wmma_f32_16x16x32_bf16(false, a, false, b,
                                                 (short)0, c, false, false);
}

// Operand where the lane dim (A rows / B cols) is the outer memory dimension
// and K is contiguous: pairs (k,k+1) -> one dword load (global or ds).
__device__ __forceinline__ v16bf ld_rowmajor_bf16(const __bf16* base, int stride,
                                                  int m, int hi) {
  union { v16bf v; uint32_t u[8]; } r;
  const __bf16* rp = base + (size_t)m * stride;
#pragma unroll
  for (int j = 0; j < 8; ++j) {
    int k0 = kidx16(2 * j, hi);               // even -> 4B aligned pair
    r.u[j] = *(const uint32_t*)(rp + k0);
  }
  return r.v;
}

__device__ __forceinline__ v16bf ld_rowmajor_f32cvt(const float* base, int stride,
                                                    int m, int hi) {
  v16bf a;
  const float* rp = base + (size_t)m * stride;
#pragma unroll
  for (int j = 0; j < 8; ++j) {
    int k0 = kidx16(2 * j, hi);
    a[2 * j]     = (__bf16)rp[k0];
    a[2 * j + 1] = (__bf16)rp[k0 + 1];
  }
  return a;
}

__device__ __forceinline__ v16bf ld_colstride_f32cvt(const float* base, int stride,
                                                     int n, int hi) {
  v16bf b;
#pragma unroll
  for (int h = 0; h < 16; ++h) {
    int k = kidx16(h, hi);
    b[h] = (__bf16)base[(size_t)k * stride + n];
  }
  return b;
}

__device__ __forceinline__ v16bf ld_colstride_bf16(const __bf16* base, int stride,
                                                   int n, int hi) {
  v16bf b;
#pragma unroll
  for (int h = 0; h < 16; ++h) {
    int k = kidx16(h, hi);
    b[h] = base[(size_t)k * stride + n];
  }
  return b;
}

// ---------------------------------------------------------------------------
// TDM: issue a 2D bf16 tile DMA global -> LDS per CDNA5 ISA D#.
//   group0: [1:0]=count=1 | lds_addr | global_addr[56:0] | type=2
//   group1: data_size=1 (2B), tensor_dim0/1, tile_dim0/1, tensor_dim0_stride
// Sizes/strides in elements (2B units). Tracked by TENSORcnt.
// This toolchain's builtin is the 6-arg form:
//   (uint32x4 g0, int32x8 g1, int32x4 g2, int32x4 g3, int32x8 pad, i32 cpol)
// ---------------------------------------------------------------------------
__device__ __forceinline__ uint32_t lds_off(const void* p) {
  // generic-address low 32 bits == wave-relative LDS byte offset
  return (uint32_t)(uintptr_t)p;
}

__device__ __forceinline__ void tdm_load_2d(uint32_t lds_byte_off, const void* gaddr,
                                            uint32_t tile_w, uint32_t tile_h,
                                            uint32_t stride0) {
  uint64_t ga = (uint64_t)(uintptr_t)gaddr;
  v4u g0;
  g0[0] = 1u;                                          // count=1 (user D#)
  g0[1] = lds_byte_off;                                // lds_addr
  g0[2] = (uint32_t)(ga & 0xFFFFFFFFu);                // global_addr[31:0]
  g0[3] = (uint32_t)((ga >> 32) & 0x01FFFFFFu) | (2u << 30);  // [56:32] | type=2
  v8i g1;
  g1[0] = (int)(1u << 16);                             // wg_mask=0, data_size=2B
  g1[1] = (int)((tile_w & 0xFFFFu) << 16);             // tensor_dim0[15:0]
  g1[2] = (int)(((tile_w >> 16) & 0xFFFFu) | ((tile_h & 0xFFFFu) << 16));
  g1[3] = (int)(((tile_h >> 16) & 0xFFFFu) | (tile_w << 16));  // tile_dim0
  g1[4] = (int)tile_h;                                 // tile_dim1 (tile_dim2=0)
  g1[5] = (int)stride0;                                // tensor_dim0_stride[31:0]
  g1[6] = 0;
  g1[7] = 0;
  v4i z4 = {0, 0, 0, 0};                               // groups 2/3 unused (2D)
  v8i z8 = {0, 0, 0, 0, 0, 0, 0, 0};
  __builtin_amdgcn_tensor_load_to_lds(g0, g1, z4, z4, z8, 0);
}

// ---------------------------------------------------------------------------
// Kernel 1: qkv = w_qkv(768x256) @ x(256x4096) per batch, scattered into
//   Q[b,h,n,d], K[b,h,n,d] (row-major d), Vt[b,h,d,n]. One wave = one 16x16
//   f32 tile via 8 WMMA K-steps.
// ---------------------------------------------------------------------------
__global__ __launch_bounds__(32) void qkv_kernel(const float* __restrict__ x,
                                                 const float* __restrict__ wqkv,
                                                 __bf16* __restrict__ Q,
                                                 __bf16* __restrict__ K,
                                                 __bf16* __restrict__ Vt) {
  const int lane = threadIdx.x, hi = lane >> 4, col = lane & 15;
  const int bid = blockIdx.x;
  const int nt = bid & 255;
  const int rt = (bid >> 8) % 48;
  const int b  = bid / (256 * 48);
  const int row0 = rt * 16, n0 = nt * 16;
  const float* X = x + (size_t)b * CCH * NPIX;

  v8f acc = {};
#pragma unroll
  for (int kc = 0; kc < 8; ++kc) {
    v16bf aw = ld_rowmajor_f32cvt(wqkv + (size_t)row0 * CCH + kc * 32, CCH, col, hi);
    v16bf bx = ld_colstride_f32cvt(X + (size_t)(kc * 32) * NPIX + n0, NPIX, col, hi);
    acc = wmma_bf16(aw, bx, acc);
  }

  const int n = n0 + col;
#pragma unroll
  for (int vg = 0; vg < 8; ++vg) {
    int r = row0 + vg + (hi << 3);
    int which = r >> 8;                 // 0:q 1:k 2:v
    int rc = r & 255;
    int h = rc >> 5, dim = rc & 31;
    size_t bh = (size_t)b * NH + h;
    __bf16 bv = (__bf16)acc[vg];
    if (which == 0)      Q[(bh * NPIX + n) * DH + dim] = bv;
    else if (which == 1) K[(bh * NPIX + n) * DH + dim] = bv;
    else                 Vt[(bh * DH + dim) * NPIX + n] = bv;
  }
}

// ---------------------------------------------------------------------------
// Kernel 2: flash attention. 256-thread WG = 8 waves x 16 queries of one
// (b,h). K/V chunks (64 keys) are TDM-DMA'd into double-buffered LDS tiles by
// wave 0 and consumed by all waves; online softmax batched per chunk; P is
// bounced through a per-wave LDS region to convert C-layout -> A-layout.
// ---------------------------------------------------------------------------
__global__ __launch_bounds__(256) void attn_kernel(const __bf16* __restrict__ Q,
                                                   const __bf16* __restrict__ K,
                                                   const __bf16* __restrict__ Vt,
                                                   __bf16* __restrict__ AO) {
  __shared__ __align__(16) __bf16 sK[2][KCH * DH];   // [key][d]
  __shared__ __align__(16) __bf16 sV[2][DH * KCH];   // [d][key]
  __shared__ __align__(16) __bf16 sP[8][16 * KCH];   // per-wave P tile

  const int tid = threadIdx.x;
  const int wave = tid >> 5;
  const int lane = tid & 31;
  const int hi = lane >> 4, col = lane & 15;

  const int bid = blockIdx.x;
  const int qblk = bid & 31;            // 32 blocks of 128 queries
  const int bh   = bid >> 5;            // b*8 + h
  const int q0 = qblk * 128 + wave * 16;

  const __bf16* Qp = Q  + ((size_t)bh * NPIX + q0) * DH;
  const __bf16* Kp = K  + (size_t)bh * NPIX * DH;
  const __bf16* Vp = Vt + (size_t)bh * DH * NPIX;

  const v16bf aq = ld_rowmajor_bf16(Qp, DH, col, hi);   // 16x32 Q tile (A op)

  v8f o0 = {}, o1 = {};
  float mrow[8], lrow[8];
#pragma unroll
  for (int i = 0; i < 8; ++i) { mrow[i] = -3.0e38f; lrow[i] = 0.0f; }
  const float scale = 0.17677669529663687f;             // d^-0.5
  const v8f zero = {};

  // Preload chunk 0 (K: 64x32 contiguous rows; V: 32 rows x 64 keys, stride N)
  if (wave == 0) {
    tdm_load_2d(lds_off(&sK[0][0]), Kp, DH, KCH, DH);
    tdm_load_2d(lds_off(&sV[0][0]), Vp, KCH, DH, NPIX);
    __builtin_amdgcn_s_wait_tensorcnt(0);
  }
  __syncthreads();

  for (int kb = 0; kb < NPIX / KCH; ++kb) {
    const int cur = kb & 1;
    if (wave == 0 && kb + 1 < NPIX / KCH) {     // prefetch next chunk
      tdm_load_2d(lds_off(&sK[cur ^ 1][0]), Kp + (size_t)(kb + 1) * KCH * DH, DH, KCH, DH);
      tdm_load_2d(lds_off(&sV[cur ^ 1][0]), Vp + (size_t)(kb + 1) * KCH, KCH, DH, NPIX);
    }
    const __bf16* lK = sK[cur];
    const __bf16* lV = sV[cur];
    __bf16* lP = sP[wave];

    // scores: 4 tiles of 16 keys
    v8f st[4];
#pragma unroll
    for (int t = 0; t < 4; ++t) {
      v16bf bk = ld_rowmajor_bf16(lK + (size_t)(t * 16) * DH, DH, col, hi);
      st[t] = wmma_bf16(aq, bk, zero);
    }

    // online softmax, one stats update per 64 keys
#pragma unroll
    for (int vg = 0; vg < 8; ++vg) {
      float a0 = st[0][vg] * scale, a1 = st[1][vg] * scale;
      float a2 = st[2][vg] * scale, a3 = st[3][vg] * scale;
      float mx = fmaxf(fmaxf(a0, a1), fmaxf(a2, a3));
#pragma unroll
      for (int off = 1; off < 16; off <<= 1) mx = fmaxf(mx, __shfl_xor(mx, off, 32));
      float mnew  = fmaxf(mrow[vg], mx);
      float alpha = __expf(mrow[vg] - mnew);
      float p0 = __expf(a0 - mnew), p1 = __expf(a1 - mnew);
      float p2 = __expf(a2 - mnew), p3 = __expf(a3 - mnew);
      float ls = (p0 + p1) + (p2 + p3);
#pragma unroll
      for (int off = 1; off < 16; off <<= 1) ls += __shfl_xor(ls, off, 32);
      lrow[vg] = lrow[vg] * alpha + ls;
      mrow[vg] = mnew;
      o0[vg] *= alpha;
      o1[vg] *= alpha;
      int r = vg + (hi << 3);
      lP[r * KCH + 0 * 16 + col] = (__bf16)p0;
      lP[r * KCH + 1 * 16 + col] = (__bf16)p1;
      lP[r * KCH + 2 * 16 + col] = (__bf16)p2;
      lP[r * KCH + 3 * 16 + col] = (__bf16)p3;
    }
    asm volatile("s_wait_dscnt 0" ::: "memory");   // cross-lane LDS RAW on lP

    // O += P(16x64) x V(64x32): 2 K-steps x 2 dim tiles
#pragma unroll
    for (int kc = 0; kc < KCH / 32; ++kc) {
      v16bf ap  = ld_rowmajor_bf16(lP + kc * 32, KCH, col, hi);
      v16bf bv0 = ld_rowmajor_bf16(lV + kc * 32, KCH, col, hi);
      v16bf bv1 = ld_rowmajor_bf16(lV + (size_t)16 * KCH + kc * 32, KCH, col, hi);
      o0 = wmma_bf16(ap, bv0, o0);
      o1 = wmma_bf16(ap, bv1, o1);
    }

    if (wave == 0) __builtin_amdgcn_s_wait_tensorcnt(0);  // next buffer landed
    asm volatile("s_wait_dscnt 0" ::: "memory");          // reads of cur done
    __syncthreads();
  }

  const int b = bh >> 3, h = bh & 7;
#pragma unroll
  for (int vg = 0; vg < 8; ++vg) {
    float inv = 1.0f / lrow[vg];
    int r = vg + (hi << 3);
    int n = q0 + r;
    size_t base = ((size_t)b * CCH + h * DH) * NPIX + n;
    AO[base + (size_t)col * NPIX]        = (__bf16)(o0[vg] * inv);
    AO[base + (size_t)(col + 16) * NPIX] = (__bf16)(o1[vg] * inv);
  }
}

// ---------------------------------------------------------------------------
// Kernel 3: out = w_proj(256x256) @ AO(256x4096) + b_proj, fp32 to d_out.
// ---------------------------------------------------------------------------
__global__ __launch_bounds__(32) void proj_kernel(const __bf16* __restrict__ AO,
                                                  const float* __restrict__ wproj,
                                                  const float* __restrict__ bproj,
                                                  float* __restrict__ out) {
  const int lane = threadIdx.x, hi = lane >> 4, col = lane & 15;
  const int bid = blockIdx.x;
  const int nt = bid & 255;
  const int rt = (bid >> 8) & 15;
  const int b  = bid >> 12;
  const int row0 = rt * 16, n0 = nt * 16;
  const __bf16* A = AO + (size_t)b * CCH * NPIX;

  v8f acc = {};
#pragma unroll
  for (int kc = 0; kc < 8; ++kc) {
    v16bf aw = ld_rowmajor_f32cvt(wproj + (size_t)row0 * CCH + kc * 32, CCH, col, hi);
    v16bf bx = ld_colstride_bf16(A + (size_t)(kc * 32) * NPIX + n0, NPIX, col, hi);
    acc = wmma_bf16(aw, bx, acc);
  }

  const int n = n0 + col;
#pragma unroll
  for (int vg = 0; vg < 8; ++vg) {
    int r = row0 + vg + (hi << 3);
    out[((size_t)b * CCH + r) * NPIX + n] = acc[vg] + bproj[r];
  }
}

// ---------------------------------------------------------------------------
extern "C" void kernel_launch(void* const* d_in, const int* in_sizes, int n_in,
                              void* d_out, int out_size, void* d_ws, size_t ws_size,
                              hipStream_t stream) {
  const float* x     = (const float*)d_in[0];  // (2,256,64,64)
  const float* wqkv  = (const float*)d_in[1];  // (768,256)
  const float* wproj = (const float*)d_in[2];  // (256,256)
  const float* bproj = (const float*)d_in[3];  // (256,)
  float* out = (float*)d_out;                  // (2,256,64,64)

  const size_t per = (size_t)BATCH * NH * NPIX * DH;   // 2,097,152 bf16 each
  __bf16* Q  = (__bf16*)d_ws;
  __bf16* K  = Q + per;
  __bf16* Vt = K + per;
  __bf16* AO = Vt + per;                               // 16 MB total

  qkv_kernel<<<BATCH * 48 * 256, 32, 0, stream>>>(x, wqkv, Q, K, Vt);
  attn_kernel<<<BATCH * NH * (NPIX / 128), 256, 0, stream>>>(Q, K, Vt, AO);
  proj_kernel<<<BATCH * 16 * 256, 32, 0, stream>>>(AO, wproj, bproj, out);
}
